// RegressionHead_42691974922481
// MI455X (gfx1250) — matrix operand order repaired and checked by reading, not compiled, not run on hardware
//
#include <hip/hip_runtime.h>
#include <math.h>

typedef __bf16 bf16_t;
typedef __attribute__((ext_vector_type(16))) __bf16 bf16x16;
typedef __attribute__((ext_vector_type(8)))  float  f32x8;

union FragU { uint4 u[2]; bf16x16 v; };

__device__ inline f32x8 wmma_bf16(bf16x16 a, bf16x16 b, f32x8 c) {
  return __builtin_amdgcn_wmma_f32_16x16x32_bf16(false, a, false, b, (short)0, c, false, false);
}
__device__ inline f32x8 zero8() { f32x8 z = {0.f,0.f,0.f,0.f,0.f,0.f,0.f,0.f}; return z; }
__device__ inline float sigm(float x) { return 1.0f / (1.0f + __expf(-x)); }
// branch-free tanh: saturates to +/-1 correctly (exp->inf => 1, exp->0 => -1)
__device__ inline float ftanh(float x) { return 1.0f - 2.0f / (__expf(2.0f * x) + 1.0f); }

// A fragment (16x32 bf16): per-lane elements [0..7] and [16..23] relative to (kb + khalfA)
__device__ inline bf16x16 load_a_bf(const bf16_t* p) {
  FragU f;
  f.u[0] = *reinterpret_cast<const uint4*>(p);
  f.u[1] = *reinterpret_cast<const uint4*>(p + 16);
  return f.v;
}
__device__ inline bf16x16 load_a_f32(const float* p) {
  bf16x16 r;
#pragma unroll
  for (int i = 0; i < 8; ++i) r[i] = (bf16_t)p[i];
#pragma unroll
  for (int i = 0; i < 8; ++i) r[i + 8] = (bf16_t)p[16 + i];
  return r;
}
// B fragment (32x16 bf16 from Wt[N,K] row-major): per-lane elements [0..7] and [16..23] relative to (kb + khalfB)
__device__ inline bf16x16 load_b_bf(const bf16_t* p) {
  FragU f;
  f.u[0] = *reinterpret_cast<const uint4*>(p);
  f.u[1] = *reinterpret_cast<const uint4*>(p + 16);
  return f.v;
}

// ---------------- pooling: mean over 7x7 per (b,c); one wave per output ----------
__global__ __launch_bounds__(256) void pool_kernel(const float* __restrict__ in,
                                                   bf16_t* __restrict__ out, int nOut) {
  int w = (blockIdx.x * blockDim.x + threadIdx.x) >> 5;
  int lane = threadIdx.x & 31;
  if (w >= nOut) return;
  const float* p = in + (long)w * 49;
  float s = p[lane];
  if (lane + 32 < 49) s += p[lane + 32];
#pragma unroll
  for (int d = 16; d > 0; d >>= 1) s += __shfl_down(s, d, 32);
  if (lane == 0) out[w] = (bf16_t)(s * (1.0f / 49.0f));
}

// ---------------- weight prep: W[K,N] f32 -> Wt[N,K] bf16 -----------------------
__global__ void transpose_to_bf16(const float* __restrict__ W, bf16_t* __restrict__ Wt,
                                  int K, int N) {
  long idx = (long)blockIdx.x * blockDim.x + threadIdx.x;
  if (idx >= (long)K * N) return;
  int n = (int)(idx / K);
  int k = (int)(idx % K);
  Wt[idx] = (bf16_t)W[(long)k * N + n];
}

__global__ void bias_sum_kernel(const float* __restrict__ a, const float* __restrict__ b,
                                float* __restrict__ o, int n) {
  int i = blockIdx.x * blockDim.x + threadIdx.x;
  if (i < n) o[i] = a[i] + b[i];
}

// sequT[(b*64+j)*32 + i] = feat[b*2048 + i*64 + j]
__global__ void sequT_kernel(const bf16_t* __restrict__ feat, bf16_t* __restrict__ sequT) {
  int idx = blockIdx.x * blockDim.x + threadIdx.x;
  if (idx >= 512 * 64 * 32) return;
  int i = idx & 31;
  int j = (idx >> 5) & 63;
  int b = idx >> 11;
  sequT[idx] = feat[(long)b * 2048 + i * 64 + j];
}

// ---------------- generic WMMA GEMM: C[M,N] = act(A[M,K] @ Wt[N,K]^T + bias) ----
// ACT: 0=none 1=relu ; AF32: A is f32 (converted in-register) else bf16
template <int ACT, bool AF32>
__global__ __launch_bounds__(256) void gemm_kernel(
    const void* __restrict__ Aptr, long lda, long aoff,
    const bf16_t* __restrict__ Wt, const float* __restrict__ bias,
    float* __restrict__ C, bf16_t* __restrict__ Cbf, long ldc,
    int M, int N, int K) {
  int gw = blockIdx.x * 8 + (threadIdx.x >> 5);
  int lane = threadIdx.x & 31;
  int nstr = N >> 6;
  int mi = gw / nstr, nj = gw % nstr;
  if (mi >= (M >> 4)) return;
  int halfsel = lane >> 4;
  int l15 = lane & 15;
  int khalfA = halfsel * 8, khalfB = halfsel * 8;
  long arow = (long)(mi * 16 + l15) * lda + aoff;
  f32x8 acc0 = zero8(), acc1 = zero8(), acc2 = zero8(), acc3 = zero8();
  for (int kb = 0; kb < K; kb += 32) {
    bf16x16 a;
    if (AF32) a = load_a_f32(reinterpret_cast<const float*>(Aptr) + arow + kb + khalfA);
    else      a = load_a_bf(reinterpret_cast<const bf16_t*>(Aptr) + arow + kb + khalfA);
    int c0 = nj * 64 + l15;
    acc0 = wmma_bf16(a, load_b_bf(Wt + (long)(c0)      * K + kb + khalfB), acc0);
    acc1 = wmma_bf16(a, load_b_bf(Wt + (long)(c0 + 16) * K + kb + khalfB), acc1);
    acc2 = wmma_bf16(a, load_b_bf(Wt + (long)(c0 + 32) * K + kb + khalfB), acc2);
    acc3 = wmma_bf16(a, load_b_bf(Wt + (long)(c0 + 48) * K + kb + khalfB), acc3);
  }
  int rbase = halfsel * 8;
#pragma unroll
  for (int j2 = 0; j2 < 4; ++j2) {
    int col = nj * 64 + j2 * 16 + l15;
    float bv = bias ? bias[col] : 0.0f;
    f32x8 acc = (j2 == 0) ? acc0 : (j2 == 1) ? acc1 : (j2 == 2) ? acc2 : acc3;
#pragma unroll
    for (int r = 0; r < 8; ++r) {
      float v = acc[r] + bv;
      if (ACT == 1) v = fmaxf(v, 0.0f);
      long off = (long)(mi * 16 + rbase + r) * ldc + col;
      if (C)   C[off] = v;
      if (Cbf) Cbf[off] = (bf16_t)v;
    }
  }
}

// ---------------- fused attention score: s = tanh(lin@w3 + b3) @ w4 + b4 --------
__global__ __launch_bounds__(256) void attn_score_kernel(
    const float* __restrict__ lin, const bf16_t* __restrict__ w3T,
    const float* __restrict__ b3, const float* __restrict__ w4,
    const float* __restrict__ b4, float* __restrict__ s_out, int M2) {
  int gw = blockIdx.x * 8 + (threadIdx.x >> 5);
  int lane = threadIdx.x & 31;
  if (gw >= (M2 >> 4)) return;
  int halfsel = lane >> 4, l15 = lane & 15;
  int khalfA = halfsel * 8, khalfB = halfsel * 8;
  long arow = (long)(gw * 16 + l15) * 64;
  f32x8 acc0 = zero8(), acc1 = zero8(), acc2 = zero8(), acc3 = zero8();
  for (int kb = 0; kb < 64; kb += 32) {
    bf16x16 a = load_a_f32(lin + arow + kb + khalfA);
    acc0 = wmma_bf16(a, load_b_bf(w3T + (long)(l15)      * 64 + kb + khalfB), acc0);
    acc1 = wmma_bf16(a, load_b_bf(w3T + (long)(l15 + 16) * 64 + kb + khalfB), acc1);
    acc2 = wmma_bf16(a, load_b_bf(w3T + (long)(l15 + 32) * 64 + kb + khalfB), acc2);
    acc3 = wmma_bf16(a, load_b_bf(w3T + (long)(l15 + 48) * 64 + kb + khalfB), acc3);
  }
  float part[8];
#pragma unroll
  for (int r = 0; r < 8; ++r) part[r] = 0.0f;
#pragma unroll
  for (int j2 = 0; j2 < 4; ++j2) {
    int col = j2 * 16 + l15;
    float b3v = b3[col], w4v = w4[col];
    f32x8 acc = (j2 == 0) ? acc0 : (j2 == 1) ? acc1 : (j2 == 2) ? acc2 : acc3;
#pragma unroll
    for (int r = 0; r < 8; ++r) part[r] += ftanh(acc[r] + b3v) * w4v;
  }
#pragma unroll
  for (int r = 0; r < 8; ++r) {
    float p = part[r];
#pragma unroll
    for (int m = 1; m < 16; m <<= 1) p += __shfl_xor(p, m, 32);
    part[r] = p;
  }
  if (l15 == 0) {
    int rbase = halfsel * 8;
    float b4v = b4[0];
#pragma unroll
    for (int r = 0; r < 8; ++r) s_out[gw * 16 + rbase + r] = part[r] + b4v;
  }
}

__global__ void smax_kernel(const float* __restrict__ s, float* __restrict__ smax,
                            int NH, int T) {
  int n = blockIdx.x * blockDim.x + threadIdx.x;
  if (n >= NH) return;
  float m = -3.4e38f;
  for (int t = 0; t < T; ++t) m = fmaxf(m, s[(long)n * T + t]);
  smax[n] = m;
}

// causal-prefix softmax weighted cumsum: one wave per head
__global__ __launch_bounds__(256) void cumattn_kernel(
    const float* __restrict__ lin, const float* __restrict__ s,
    const float* __restrict__ smax, float* __restrict__ ctx, int NH, int T) {
  int n = (blockIdx.x * blockDim.x + threadIdx.x) >> 5;
  int lane = threadIdx.x & 31;
  if (n >= NH) return;
  float mx = smax[n];
  float num0 = 0.f, num1 = 0.f, den = 0.f;
  const float* x = lin + (long)n * T * 64;
  float* c = ctx + (long)n * T * 64;
  const float* sp = s + (long)n * T;
  for (int t = 0; t < T; ++t) {
    float e = __expf(sp[t] - mx);
    den += e;
    num0 += e * x[t * 64 + lane];
    num1 += e * x[t * 64 + lane + 32];
    float inv = 1.0f / den;
    c[t * 64 + lane] = num0 * inv;
    c[t * 64 + lane + 32] = num1 * inv;
  }
}

// ---------------- fused BiLSTM recurrence ---------------------------------------
// grid: (16 batch-slices of 32, 2 dirs, 2 nets[lr,ud]); 256 threads (8 waves).
// gates[32,1024] = h_prev[32,256]@whh + x_t[32,Dx]@wih + bias; h dbl-buffered LDS.
__global__ __launch_bounds__(256) void lstm_kernel(
    const bf16_t* __restrict__ feat, const bf16_t* __restrict__ sequT,
    const bf16_t* __restrict__ lr_hfT, const bf16_t* __restrict__ lr_hbT,
    const bf16_t* __restrict__ ud_hfT, const bf16_t* __restrict__ ud_hbT,
    const bf16_t* __restrict__ lr_wfT, const bf16_t* __restrict__ lr_wbT,
    const bf16_t* __restrict__ ud_wfT, const bf16_t* __restrict__ ud_wbT,
    const float* __restrict__ lr_bF, const float* __restrict__ lr_bB,
    const float* __restrict__ ud_bF, const float* __restrict__ ud_bB,
    float* __restrict__ out_lr, float* __restrict__ out_ud) {
  __shared__ bf16_t hbuf[2][32 * 256];
  __shared__ float  cbuf[32 * 256];
  const int z = blockIdx.z, dir = blockIdx.y;
  const int b0 = blockIdx.x * 32;
  const int T = z ? 64 : 32;
  const int Dx = z ? 32 : 64;
  const bf16_t* xb   = z ? sequT : feat;
  const bf16_t* whhT = z ? (dir ? ud_hbT : ud_hfT) : (dir ? lr_hbT : lr_hfT);
  const bf16_t* wihT = z ? (dir ? ud_wbT : ud_wfT) : (dir ? lr_wbT : lr_wfT);
  const float*  bias = z ? (dir ? ud_bB : ud_bF) : (dir ? lr_bB : lr_bF);
  float* out = z ? out_ud : out_lr;
  const int dofs = dir ? 256 : 0;

  for (int i = threadIdx.x; i < 32 * 256; i += 256) {
    hbuf[0][i] = (bf16_t)0.0f;
    cbuf[i] = 0.0f;
  }
  __syncthreads();

  const int w = threadIdx.x >> 5, lane = threadIdx.x & 31;
  const int mt = w & 1;            // row tile (0..1)
  const int jt0 = w >> 1;          // 0..3 ; owned j-tiles: jt0 + 4*u
  const int halfsel = lane >> 4, l15 = lane & 15;
  const int khalfA = halfsel * 8, khalfB = halfsel * 8;
  const int rowm = mt * 16 + l15;  // A-matrix row within slice
  const int rbase = halfsel * 8;

  for (int sstep = 0; sstep < T; ++sstep) {
    const bf16_t* hread = hbuf[sstep & 1];
    bf16_t* hwrite = hbuf[(sstep + 1) & 1];
    const int tx = dir ? (T - 1 - sstep) : sstep;
    const bf16_t* xrow = xb + (long)(b0 + rowm) * 2048 + tx * Dx;
    const bf16_t* hrow = hread + rowm * 256;
#pragma unroll 1
    for (int u = 0; u < 4; ++u) {
      const int jt = jt0 + 4 * u;
      const int coln = jt * 16 + l15;
      f32x8 acc0 = zero8(), acc1 = zero8(), acc2 = zero8(), acc3 = zero8();
#pragma unroll
      for (int kb = 0; kb < 256; kb += 32) {
        bf16x16 a = load_a_bf(hrow + kb + khalfA);
        acc0 = wmma_bf16(a, load_b_bf(whhT + (long)(coln)       * 256 + kb + khalfB), acc0);
        acc1 = wmma_bf16(a, load_b_bf(whhT + (long)(coln + 256) * 256 + kb + khalfB), acc1);
        acc2 = wmma_bf16(a, load_b_bf(whhT + (long)(coln + 512) * 256 + kb + khalfB), acc2);
        acc3 = wmma_bf16(a, load_b_bf(whhT + (long)(coln + 768) * 256 + kb + khalfB), acc3);
      }
      for (int kb = 0; kb < Dx; kb += 32) {
        bf16x16 a = load_a_bf(xrow + kb + khalfA);
        acc0 = wmma_bf16(a, load_b_bf(wihT + (long)(coln)       * Dx + kb + khalfB), acc0);
        acc1 = wmma_bf16(a, load_b_bf(wihT + (long)(coln + 256) * Dx + kb + khalfB), acc1);
        acc2 = wmma_bf16(a, load_b_bf(wihT + (long)(coln + 512) * Dx + kb + khalfB), acc2);
        acc3 = wmma_bf16(a, load_b_bf(wihT + (long)(coln + 768) * Dx + kb + khalfB), acc3);
      }
      const float bi = bias[coln], bff = bias[256 + coln];
      const float bg = bias[512 + coln], bo = bias[768 + coln];
#pragma unroll
      for (int r = 0; r < 8; ++r) {
        const int m = mt * 16 + rbase + r;
        float iv = sigm(acc0[r] + bi);
        float fv = sigm(acc1[r] + bff);
        float gv = ftanh(acc2[r] + bg);
        float ov = sigm(acc3[r] + bo);
        float cn = fv * cbuf[m * 256 + coln] + iv * gv;
        cbuf[m * 256 + coln] = cn;
        float hv = ov * ftanh(cn);
        hwrite[m * 256 + coln] = (bf16_t)hv;
        out[((long)(b0 + m) * T + tx) * 512 + dofs + coln] = hv;
      }
    }
    __syncthreads();
  }
}

// ---------------- residual + layernorm on last timestep rows --------------------
__global__ __launch_bounds__(256) void ln_kernel(
    const float* __restrict__ xbase, long xrowstride, long xoff,
    const float* __restrict__ ctx2, const float* __restrict__ g,
    const float* __restrict__ bb, float* __restrict__ outcat, int colofs) {
  int bi = (blockIdx.x * blockDim.x + threadIdx.x) >> 5;
  int lane = threadIdx.x & 31;
  if (bi >= 512) return;
  const float* xr = xbase + (long)bi * xrowstride + xoff;
  const float* cr = ctx2 + (long)bi * 512;
  float v[16];
  float sum = 0.f;
#pragma unroll
  for (int i = 0; i < 16; ++i) {
    v[i] = xr[lane + 32 * i] + cr[lane + 32 * i];
    sum += v[i];
  }
#pragma unroll
  for (int d = 16; d > 0; d >>= 1) sum += __shfl_xor(sum, d, 32);
  float mean = sum * (1.0f / 512.0f);
  float var = 0.f;
#pragma unroll
  for (int i = 0; i < 16; ++i) {
    float t = v[i] - mean;
    var += t * t;
  }
#pragma unroll
  for (int d = 16; d > 0; d >>= 1) var += __shfl_xor(var, d, 32);
  var *= (1.0f / 512.0f);
  float rs = rsqrtf(var + 1e-5f);
#pragma unroll
  for (int i = 0; i < 16; ++i) {
    int c = lane + 32 * i;
    outcat[(long)bi * 1024 + colofs + c] = (v[i] - mean) * rs * g[c] + bb[c];
  }
}

// ---------------- output heads: xy = out@xyw+b ; wpqr normalized ----------------
__global__ __launch_bounds__(256) void head_kernel(
    const float* __restrict__ outcat, const float* __restrict__ xyw,
    const float* __restrict__ xyb, const float* __restrict__ qw,
    const float* __restrict__ qb, float* __restrict__ dout) {
  int bi = (blockIdx.x * blockDim.x + threadIdx.x) >> 5;
  int lane = threadIdx.x & 31;
  if (bi >= 512) return;
  float a[7];
#pragma unroll
  for (int j = 0; j < 7; ++j) a[j] = 0.f;
  const float* row = outcat + (long)bi * 1024;
  for (int k = lane; k < 1024; k += 32) {
    float v = row[k];
    a[0] += v * xyw[k * 3 + 0];
    a[1] += v * xyw[k * 3 + 1];
    a[2] += v * xyw[k * 3 + 2];
    a[3] += v * qw[k * 4 + 0];
    a[4] += v * qw[k * 4 + 1];
    a[5] += v * qw[k * 4 + 2];
    a[6] += v * qw[k * 4 + 3];
  }
#pragma unroll
  for (int j = 0; j < 7; ++j) {
#pragma unroll
    for (int d = 16; d > 0; d >>= 1) a[j] += __shfl_xor(a[j], d, 32);
  }
  if (lane == 0) {
    dout[bi * 3 + 0] = a[0] + xyb[0];
    dout[bi * 3 + 1] = a[1] + xyb[1];
    dout[bi * 3 + 2] = a[2] + xyb[2];
    float w0 = a[3] + qb[0], w1 = a[4] + qb[1], w2 = a[5] + qb[2], w3 = a[6] + qb[3];
    float nn = fmaxf(sqrtf(w0 * w0 + w1 * w1 + w2 * w2 + w3 * w3), 1e-12f);
    float inv = 1.0f / nn;
    dout[1536 + bi * 4 + 0] = w0 * inv;
    dout[1536 + bi * 4 + 1] = w1 * inv;
    dout[1536 + bi * 4 + 2] = w2 * inv;
    dout[1536 + bi * 4 + 3] = w3 * inv;
  }
}

// ================================ host side =====================================
extern "C" void kernel_launch(void* const* d_in, const int* in_sizes, int n_in,
                              void* d_out, int out_size, void* d_ws, size_t ws_size,
                              hipStream_t stream) {
  const float* input  = (const float*)d_in[0];
  const float* fc_w   = (const float*)d_in[1];
  const float* fc_b   = (const float*)d_in[2];
  const float* lr_wf  = (const float*)d_in[3];
  const float* lr_hf  = (const float*)d_in[4];
  const float* lr_bif = (const float*)d_in[5];
  const float* lr_bhf = (const float*)d_in[6];
  const float* lr_wb  = (const float*)d_in[7];
  const float* lr_hb  = (const float*)d_in[8];
  const float* lr_bib = (const float*)d_in[9];
  const float* lr_bhb = (const float*)d_in[10];
  const float* ud_wf  = (const float*)d_in[11];
  const float* ud_hf  = (const float*)d_in[12];
  const float* ud_bif = (const float*)d_in[13];
  const float* ud_bhf = (const float*)d_in[14];
  const float* ud_wb  = (const float*)d_in[15];
  const float* ud_hb  = (const float*)d_in[16];
  const float* ud_bib = (const float*)d_in[17];
  const float* ud_bhb = (const float*)d_in[18];
  const float* a3w    = (const float*)d_in[19];
  const float* a3b    = (const float*)d_in[20];
  const float* a4w    = (const float*)d_in[21];
  const float* a4b    = (const float*)d_in[22];
  const float* mhlw   = (const float*)d_in[23];
  const float* mhlb   = (const float*)d_in[24];
  const float* mflw   = (const float*)d_in[25];
  const float* mflb   = (const float*)d_in[26];
  const float* lng    = (const float*)d_in[27];
  const float* lnbp   = (const float*)d_in[28];
  const float* xyw    = (const float*)d_in[29];
  const float* xyb    = (const float*)d_in[30];
  const float* qw     = (const float*)d_in[31];
  const float* qb     = (const float*)d_in[32];

  char* wsb = (char*)d_ws;
  size_t off = 0;
  auto alloc = [&](size_t bytes) -> void* {
    off = (off + 255) & ~(size_t)255;
    void* p = wsb + off;
    off += bytes;
    return p;
  };

  bf16_t* fcT    = (bf16_t*)alloc(2048UL * 1024 * 2);
  bf16_t* lr_wfT = (bf16_t*)alloc(1024UL * 64 * 2);
  bf16_t* lr_wbT = (bf16_t*)alloc(1024UL * 64 * 2);
  bf16_t* ud_wfT = (bf16_t*)alloc(1024UL * 32 * 2);
  bf16_t* ud_wbT = (bf16_t*)alloc(1024UL * 32 * 2);
  bf16_t* lr_hfT = (bf16_t*)alloc(1024UL * 256 * 2);
  bf16_t* lr_hbT = (bf16_t*)alloc(1024UL * 256 * 2);
  bf16_t* ud_hfT = (bf16_t*)alloc(1024UL * 256 * 2);
  bf16_t* ud_hbT = (bf16_t*)alloc(1024UL * 256 * 2);
  bf16_t* mhlT   = (bf16_t*)alloc(512UL * 512 * 2);
  bf16_t* mflT   = (bf16_t*)alloc(512UL * 512 * 2);
  bf16_t* a3T    = (bf16_t*)alloc(64UL * 64 * 2);
  float* lr_bF   = (float*)alloc(1024UL * 4);
  float* lr_bB   = (float*)alloc(1024UL * 4);
  float* ud_bF   = (float*)alloc(1024UL * 4);
  float* ud_bB   = (float*)alloc(1024UL * 4);
  bf16_t* pooled = (bf16_t*)alloc(512UL * 1024 * 2);
  bf16_t* featb  = (bf16_t*)alloc(512UL * 2048 * 2);
  bf16_t* seqT   = (bf16_t*)alloc(512UL * 64 * 32 * 2);
  float* out_lr  = (float*)alloc(512UL * 32 * 512 * 4);
  float* out_ud  = (float*)alloc(512UL * 64 * 512 * 4);
  float* lin     = (float*)alloc(32768UL * 512 * 4);   // shared lr/ud
  float* ctx     = (float*)alloc(32768UL * 512 * 4);   // shared lr/ud
  float* svec    = (float*)alloc(262144UL * 4);
  float* smax    = (float*)alloc(4096UL * 4);
  float* ctx2    = (float*)alloc(512UL * 512 * 4);
  float* outcat  = (float*)alloc(512UL * 1024 * 4);
  (void)ws_size; (void)in_sizes; (void)n_in; (void)out_size;

  auto TT = [&](const float* W, bf16_t* Wt, int K, int N) {
    long tot = (long)K * N;
    transpose_to_bf16<<<(int)((tot + 255) / 256), 256, 0, stream>>>(W, Wt, K, N);
  };
  TT(fc_w, fcT, 1024, 2048);
  TT(lr_wf, lr_wfT, 64, 1024);
  TT(lr_wb, lr_wbT, 64, 1024);
  TT(ud_wf, ud_wfT, 32, 1024);
  TT(ud_wb, ud_wbT, 32, 1024);
  TT(lr_hf, lr_hfT, 256, 1024);
  TT(lr_hb, lr_hbT, 256, 1024);
  TT(ud_hf, ud_hfT, 256, 1024);
  TT(ud_hb, ud_hbT, 256, 1024);
  TT(mhlw, mhlT, 512, 512);
  TT(mflw, mflT, 512, 512);
  TT(a3w, a3T, 64, 64);
  bias_sum_kernel<<<4, 256, 0, stream>>>(lr_bif, lr_bhf, lr_bF, 1024);
  bias_sum_kernel<<<4, 256, 0, stream>>>(lr_bib, lr_bhb, lr_bB, 1024);
  bias_sum_kernel<<<4, 256, 0, stream>>>(ud_bif, ud_bhf, ud_bF, 1024);
  bias_sum_kernel<<<4, 256, 0, stream>>>(ud_bib, ud_bhb, ud_bB, 1024);

  // pool: 512*1024 outputs, one wave each (8 per block)
  pool_kernel<<<512 * 1024 / 8, 256, 0, stream>>>(input, pooled, 512 * 1024);

  // FC + ReLU -> feat bf16 [512,2048]  (strips = 32*32 = 1024)
  gemm_kernel<1, false><<<128, 256, 0, stream>>>(pooled, 1024, 0, fcT, fc_b,
                                                 nullptr, featb, 2048, 512, 2048, 1024);
  sequT_kernel<<<(512 * 64 * 32 + 255) / 256, 256, 0, stream>>>(featb, seqT);

  // all four BiLSTM recurrences in one launch
  lstm_kernel<<<dim3(16, 2, 2), 256, 0, stream>>>(
      featb, seqT, lr_hfT, lr_hbT, ud_hfT, ud_hbT, lr_wfT, lr_wbT, ud_wfT, ud_wbT,
      lr_bF, lr_bB, ud_bF, ud_bB, out_lr, out_ud);

  // ---- MHA phase for lr (T=32) ----
  {
    const int T = 32, NH = 4096, M = 512 * T, M2 = NH * T;
    gemm_kernel<0, true><<<(M / 16) * 8 / 8, 256, 0, stream>>>(
        out_lr, 512, 0, mhlT, mhlb, lin, nullptr, 512, M, 512, 512);
    attn_score_kernel<<<(M2 / 16) / 8, 256, 0, stream>>>(lin, a3T, a3b, a4w, a4b, svec, M2);
    smax_kernel<<<NH / 256, 256, 0, stream>>>(svec, smax, NH, T);
    cumattn_kernel<<<NH / 8, 256, 0, stream>>>(lin, svec, smax, ctx, NH, T);
    gemm_kernel<0, true><<<(512 / 16) * 8 / 8, 256, 0, stream>>>(
        ctx, (long)T * 512, (long)(T - 1) * 512, mflT, mflb, ctx2, nullptr, 512, 512, 512, 512);
    ln_kernel<<<512 / 8, 256, 0, stream>>>(out_lr, (long)T * 512, (long)(T - 1) * 512,
                                           ctx2, lng, lnbp, outcat, 0);
  }
  // ---- MHA phase for ud (T=64) ----
  {
    const int T = 64, NH = 4096, M = 512 * T, M2 = NH * T;
    gemm_kernel<0, true><<<(M / 16) * 8 / 8, 256, 0, stream>>>(
        out_ud, 512, 0, mhlT, mhlb, lin, nullptr, 512, M, 512, 512);
    attn_score_kernel<<<(M2 / 16) / 8, 256, 0, stream>>>(lin, a3T, a3b, a4w, a4b, svec, M2);
    smax_kernel<<<NH / 256, 256, 0, stream>>>(svec, smax, NH, T);
    cumattn_kernel<<<NH / 8, 256, 0, stream>>>(lin, svec, smax, ctx, NH, T);
    gemm_kernel<0, true><<<(512 / 16) * 8 / 8, 256, 0, stream>>>(
        ctx, (long)T * 512, (long)(T - 1) * 512, mflT, mflb, ctx2, nullptr, 512, 512, 512, 512);
    ln_kernel<<<512 / 8, 256, 0, stream>>>(out_ud, (long)T * 512, (long)(T - 1) * 512,
                                           ctx2, lng, lnbp, outcat, 512);
  }

  head_kernel<<<512 / 8, 256, 0, stream>>>(outcat, xyw, xyb, qw, qb, (float*)d_out);
}
